// CARAFEFast_20864951124543
// MI455X (gfx1250) — matrix operand order repaired and checked by reading, not compiled, not run on hardware
//
#include <hip/hip_runtime.h>
#include <hip/hip_bf16.h>

// ---------------------------------------------------------------------------
// CARAFE upsampling for MI455X (gfx1250): bf16 WMMA GEMMs + async-LDS staging
// ---------------------------------------------------------------------------

typedef __attribute__((ext_vector_type(16))) __bf16 v16bf;
typedef __attribute__((ext_vector_type(8)))  float  v8f;

union Frag16 { v16bf v; unsigned int u[8]; };
union FragC  { v8f   v; float        f[8]; };

// float -> bf16 (round to nearest even), raw bits
__device__ __forceinline__ unsigned short f2bf(float f) {
    unsigned int u = __float_as_uint(f);
    u += 0x7FFFu + ((u >> 16) & 1u);
    return (unsigned short)(u >> 16);
}

// Async global->LDS 16B copy (CDNA5 GLOBAL_LOAD_ASYNC_TO_LDS_B128, ASYNCcnt).
// Low 32 bits of a generic LDS address are the LDS byte address (ISA 10.2).
__device__ __forceinline__ void async_copy16(const void* gptr, void* lptr) {
    unsigned lds = (unsigned)(unsigned long long)lptr;
    asm volatile("global_load_async_to_lds_b128 %0, %1, off"
                 :: "v"(lds), "v"(gptr) : "memory");
}
__device__ __forceinline__ void wait_async0() {
    asm volatile("s_wait_asynccnt 0x0" ::: "memory");
}

// Problem constants
#define BB     4
#define CC     128
#define HH     64
#define WW     64
#define HW     4096          // 64*64
#define CMID   64
#define NKER   100           // s*s*k*k
#define NPAD   128           // padded N for conv GEMM
#define KCONV  576           // CMID * 3 * 3

// ---------------------------------------------------------------------------
// Kernel 0: one-time weight conversion to bf16.
//   w1bf: (64,128) bf16  == w1 elementwise
//   w2bf: (128,576) bf16 == w2 rows, zero-padded for n in [100,128)
// ---------------------------------------------------------------------------
__global__ void __launch_bounds__(256)
cvt_weights_kernel(const float* __restrict__ w1, const float* __restrict__ w2,
                   unsigned short* __restrict__ w1bf,
                   unsigned short* __restrict__ w2bf)
{
    const int t = blockIdx.x * 256 + threadIdx.x;
    if (t < CMID * CC) w1bf[t] = f2bf(w1[t]);
    const int i = t - CMID * CC;
    if (i >= 0 && i < NPAD * KCONV) {
        const int n = i / KCONV;
        w2bf[i] = (n < NKER) ? f2bf(w2[i]) : (unsigned short)0;
    }
}

// ---------------------------------------------------------------------------
// Kernel 1: h = ReLU(BN(x @ w1^T)) -> bf16, layout (B, CMID, H, W)
// GEMM: M=16384, N=64, K=128. WG = 8 waves; wave (mi,ni) owns a 16x16 tile of
// a 32(pixel) x 64(cmid) block. All 4 K-chunks' fragments are loaded up front
// so the loads overlap and the 4 WMMAs issue back-to-back.
// ---------------------------------------------------------------------------
__global__ void __launch_bounds__(256)
enc_gemm_kernel(const float* __restrict__ x,
                const unsigned short* __restrict__ w1bf,
                const float* __restrict__ bg, const float* __restrict__ bb,
                const float* __restrict__ bm, const float* __restrict__ bv,
                unsigned short* __restrict__ h)
{
    __shared__ unsigned short sA[32][128];   // [pixel][k] bf16

    const int t   = threadIdx.x;
    const int m0  = blockIdx.x * 32;
    const int b   = m0 >> 12;
    const int hw0 = m0 & (HW - 1);
    const float* xb = x + (size_t)b * CC * HW;

    for (int i = t; i < 32 * 128; i += 256) {
        const int c = i >> 5, p = i & 31;
        sA[p][c] = f2bf(xb[(size_t)c * HW + hw0 + p]);
    }
    __syncthreads();

    const int wave = t >> 5;
    const int lane = t & 31;
    const int mi   = wave >> 2;
    const int ni   = wave & 3;
    const int lo16 = lane & 15;
    const bool hiL = (lane & 16) != 0;
    const int p    = mi * 16 + lo16;
    const int n    = ni * 16 + lo16;
    const int kba  = hiL ? 16 : 0;
    const int kbb  = hiL ? 16 : 0;

    // Load every fragment first (distinct registers -> overlapping loads)
    Frag16 a[4], bfr[4];
    #pragma unroll
    for (int kc = 0; kc < 4; ++kc) {
        const int k0 = kc * 32;
        const unsigned int* gb =
            (const unsigned int*)(w1bf + n * 128 + k0 + kbb);   // 16B aligned
        #pragma unroll
        for (int j = 0; j < 8; ++j) {
            const int ka = k0 + kba + j * 2;
            a[kc].u[j]   = *(const unsigned int*)&sA[p][ka];
            bfr[kc].u[j] = gb[j];
        }
    }

    FragC acc;
    #pragma unroll
    for (int j = 0; j < 8; ++j) acc.f[j] = 0.f;
    #pragma unroll
    for (int kc = 0; kc < 4; ++kc)
        acc.v = __builtin_amdgcn_wmma_f32_16x16x32_bf16(
                    false, a[kc].v, false, bfr[kc].v, (short)0, acc.v, false, false);

    const float inv = bg[n] * rsqrtf(bv[n] + 1e-5f);
    const float bet = bb[n] - bm[n] * inv;
    union { uint4 v; unsigned short s[8]; } pk;
    #pragma unroll
    for (int j = 0; j < 8; ++j) {
        float v = acc.f[j] * inv + bet;
        pk.s[j] = f2bf(v > 0.f ? v : 0.f);
    }
    const size_t off = ((size_t)(b * CMID + n)) * HW + hw0 + mi * 16 + (hiL ? 8 : 0);
    *(uint4*)(h + off) = pk.v;
}

// ---------------------------------------------------------------------------
// Kernel 2: ker = conv3x3(h, w2) as implicit GEMM. M=16384, N=128(pad), K=576.
// WG = 8 waves; each wave owns one 16-wide n-tile for a 32-pixel m-tile
// (two accumulators share each B fragment). A tile staged once in LDS;
// B fragments streamed from L2. K-loop is double-buffered: fragments for
// chunk kc+1 are issued before the WMMAs of chunk kc.
// ---------------------------------------------------------------------------
__global__ void __launch_bounds__(256)
ker_conv_kernel(const unsigned short* __restrict__ h,
                const unsigned short* __restrict__ w2bf,
                float* __restrict__ ker)
{
    __shared__ unsigned short sA[32][KCONV];  // 36 KB halo tile (bf16 copy)

    const int t   = threadIdx.x;
    const int m0  = blockIdx.x * 32;
    const int b   = m0 >> 12;
    const int hw0 = m0 & (HW - 1);
    const int y   = hw0 >> 6;
    const int x0  = hw0 & 63;                 // 0 or 32
    const unsigned short* hb = h + (size_t)b * CMID * HW;

    // stage A: coalesced over pixels (pp fastest across threads)
    for (int i = t; i < 32 * KCONV; i += 256) {
        const int pp = i & 31, k = i >> 5;
        const int c  = k / 9,  r = k - c * 9;
        const int dy = r / 3 - 1, dx = r - (r / 3) * 3 - 1;
        const int yy = y + dy, xx = x0 + pp + dx;
        unsigned short v = 0;
        if (yy >= 0 && yy < HH && xx >= 0 && xx < WW)
            v = hb[(size_t)c * HW + yy * WW + xx];
        sA[pp][k] = v;
    }
    __syncthreads();

    const int wave = t >> 5;
    const int lane = t & 31;
    const int lo16 = lane & 15;
    const bool hiL = (lane & 16) != 0;
    const int n    = wave * 16 + lo16;        // 0..127
    const int kba  = hiL ? 16 : 0;
    const int kbb  = hiL ? 16 : 0;

    const unsigned short* wrow = w2bf + n * KCONV + kbb;

    Frag16 a0[2], a1[2], bfr[2];

    // fragment loader for K-chunk kc into pipeline slot s
    auto load_frags = [&](int kc, int s) {
        const unsigned int* gb = (const unsigned int*)(wrow + kc * 32);
        #pragma unroll
        for (int j = 0; j < 8; ++j) bfr[s].u[j] = gb[j];
        #pragma unroll
        for (int j = 0; j < 8; ++j) {
            const int ka = kc * 32 + kba + j * 2;
            a0[s].u[j] = *(const unsigned int*)&sA[lo16][ka];
            a1[s].u[j] = *(const unsigned int*)&sA[16 + lo16][ka];
        }
    };

    FragC acc0, acc1;
    #pragma unroll
    for (int j = 0; j < 8; ++j) { acc0.f[j] = 0.f; acc1.f[j] = 0.f; }

    load_frags(0, 0);
    #pragma unroll
    for (int kc = 0; kc < 18; ++kc) {
        const int cur = kc & 1;
        if (kc < 17) load_frags(kc + 1, cur ^ 1);
        acc0.v = __builtin_amdgcn_wmma_f32_16x16x32_bf16(
                     false, a0[cur].v, false, bfr[cur].v, (short)0, acc0.v, false, false);
        acc1.v = __builtin_amdgcn_wmma_f32_16x16x32_bf16(
                     false, a1[cur].v, false, bfr[cur].v, (short)0, acc1.v, false, false);
    }

    if (n < NKER) {
        const size_t base = ((size_t)(b * NKER + n)) * HW + hw0 + (hiL ? 8 : 0);
        float4 q0 = { acc0.f[0], acc0.f[1], acc0.f[2], acc0.f[3] };
        float4 q1 = { acc0.f[4], acc0.f[5], acc0.f[6], acc0.f[7] };
        float4 q2 = { acc1.f[0], acc1.f[1], acc1.f[2], acc1.f[3] };
        float4 q3 = { acc1.f[4], acc1.f[5], acc1.f[6], acc1.f[7] };
        *(float4*)(ker + base)      = q0;
        *(float4*)(ker + base + 4)  = q1;
        *(float4*)(ker + base + 16) = q2;
        *(float4*)(ker + base + 20) = q3;
    }
}

// ---------------------------------------------------------------------------
// Kernel 3: softmax over the 25-tap window, in place.
// ---------------------------------------------------------------------------
__global__ void __launch_bounds__(256)
softmax25_kernel(float* __restrict__ ker)
{
    const int t  = blockIdx.x * 256 + threadIdx.x;
    const int hw = t & (HW - 1);
    const int g  = (t >> 12) & 3;
    const int b  = t >> 14;
    const size_t base = ((size_t)(b * NKER + g * 25)) * HW + hw;

    float v[25];
    float mx = -3.4e38f;
    #pragma unroll
    for (int j = 0; j < 25; ++j) {
        v[j] = ker[base + (size_t)j * HW];
        mx = fmaxf(mx, v[j]);
    }
    float s = 0.f;
    #pragma unroll
    for (int j = 0; j < 25; ++j) { v[j] = __expf(v[j] - mx); s += v[j]; }
    const float r = 1.f / s;
    #pragma unroll
    for (int j = 0; j < 25; ++j) ker[base + (size_t)j * HW] = v[j] * r;
}

// ---------------------------------------------------------------------------
// Kernel 4: weighted reassembly + pixel shuffle. Two blocks per (b, row y)
// (64 channels each) for full WGP coverage; the row's 100x64 softmax weights
// are DMA'd into LDS with GLOBAL_LOAD_ASYNC_TO_LDS_B128 and reused across
// the block's 64 channels.
// ---------------------------------------------------------------------------
__global__ void __launch_bounds__(256)
reassemble_kernel(const float* __restrict__ x,
                  const float* __restrict__ ker,
                  float* __restrict__ out)
{
    __shared__ float sK[4 * 25 * WW];        // [g][j][w] = 25.6 KB

    const int t    = threadIdx.x;
    const int half = blockIdx.x & 1;         // channel half (0..1)
    const int by   = blockIdx.x >> 1;
    const int b    = by >> 6;
    const int y    = by & 63;

    const float* kb = ker + ((size_t)b * NKER) * HW + y * WW;
    // 6400 floats = 1600 x 16B async DMA transfers
    for (int i = t; i < NKER * WW / 4; i += 256) {
        const int ch = i >> 4;               // row (channel) of 64 floats
        const int wq = (i & 15) * 4;         // 16B chunk within the row
        async_copy16(kb + (size_t)ch * HW + wq, &sK[(ch << 6) + wq]);
    }
    wait_async0();
    __syncthreads();

    const int w  = t & 63;
    const int cg = t >> 6;                   // 0..3, channel group
    const float* xb = x   + (size_t)b * CC * HW;
    float*       ob = out + (size_t)b * CC * (HW * 4);

    for (int ccI = 0; ccI < 16; ++ccI) {
        const int c = half * 64 + cg + ccI * 4;
        const float* xc = xb + (size_t)c * HW;

        float win[25];
        #pragma unroll
        for (int dy = 0; dy < 5; ++dy) {
            const int yy = y + dy - 2;
            #pragma unroll
            for (int dx = 0; dx < 5; ++dx) {
                const int xx = w + dx - 2;
                float v = 0.f;
                if (yy >= 0 && yy < HH && xx >= 0 && xx < WW)
                    v = xc[yy * WW + xx];
                win[dy * 5 + dx] = v;
            }
        }

        float acc[4] = {0.f, 0.f, 0.f, 0.f};
        #pragma unroll
        for (int j = 0; j < 25; ++j) {
            const float xv = win[j];
            #pragma unroll
            for (int g = 0; g < 4; ++g)
                acc[g] += xv * sK[(g * 25 + j) * WW + w];
        }

        float* oc = ob + (size_t)c * (HW * 4);
        #pragma unroll
        for (int sy = 0; sy < 2; ++sy) {
            float2 pr = { acc[sy * 2 + 0], acc[sy * 2 + 1] };
            *(float2*)(oc + (2 * y + sy) * (2 * WW) + 2 * w) = pr;
        }
    }
}

// ---------------------------------------------------------------------------
// Launch
// ---------------------------------------------------------------------------
extern "C" void kernel_launch(void* const* d_in, const int* in_sizes, int n_in,
                              void* d_out, int out_size, void* d_ws, size_t ws_size,
                              hipStream_t stream)
{
    (void)in_sizes; (void)n_in; (void)out_size; (void)ws_size;

    const float* x  = (const float*)d_in[0];
    const float* w1 = (const float*)d_in[1];
    const float* w2 = (const float*)d_in[2];
    const float* bg = (const float*)d_in[3];
    const float* bb = (const float*)d_in[4];
    const float* bm = (const float*)d_in[5];
    const float* bv = (const float*)d_in[6];
    float* out = (float*)d_out;

    // workspace layout (bytes):
    //   [0, 2 MB)          h bf16 (16384 x 64)
    //   [2 MB, 8.25 MB)    ker fp32 (4 x 100 x 4096)
    //   then w1bf (16 KB), w2bf (144 KB)
    char* ws = (char*)d_ws;
    unsigned short* h_ws   = (unsigned short*)ws;
    float*          ker_ws = (float*)(ws + (size_t)BB * HW * CMID * 2);
    unsigned short* w1bf   = (unsigned short*)(ws + (size_t)BB * HW * CMID * 2
                                                  + (size_t)BB * NKER * HW * 4);
    unsigned short* w2bf   = w1bf + CMID * CC;

    // K0: weight conversion (8192 + 73728 = 81920 elements)
    cvt_weights_kernel<<<320, 256, 0, stream>>>(w1, w2, w1bf, w2bf);
    // K1: 16384 pixels / 32 per WG
    enc_gemm_kernel<<<512, 256, 0, stream>>>(x, w1bf, bg, bb, bm, bv, h_ws);
    // K2: 16384 pixels / 32 per WG
    ker_conv_kernel<<<512, 256, 0, stream>>>(h_ws, w2bf, ker_ws);
    // K3: 65536 threads
    softmax25_kernel<<<256, 256, 0, stream>>>(ker_ws);
    // K4: two blocks per (b, y)
    reassemble_kernel<<<BB * HH * 2, 256, 0, stream>>>(x, ker_ws, out);
}
